// EncoderBlock_74234214744656
// MI455X (gfx1250) — matrix operand order repaired and checked by reading, not compiled, not run on hardware
//
#include <hip/hip_runtime.h>
#include <hip/hip_bf16.h>

typedef __attribute__((ext_vector_type(16))) _Float16 v16h;
typedef __attribute__((ext_vector_type(8)))  _Float16 v8h;
typedef __attribute__((ext_vector_type(8)))  float    v8f;

#define WMMA_F32_F16(A, B, C) \
    __builtin_amdgcn_wmma_f32_16x16x32_f16(false, (A), false, (B), (short)0, (C), false, false)

static constexpr int Bb  = 8;
static constexpr int Ss  = 1024;
static constexpr int Dd  = 256;
static constexpr int Hh  = 8;
static constexpr int DK  = 32;
static constexpr int HID = 1024;
static constexpr int Mrows = Bb * Ss;   // 8192

// 16 contiguous f16 -> v16h (two b128 loads)
__device__ inline v16h ld16(const _Float16* p) {
    v8h lo = *(const v8h*)p;
    v8h hi = *(const v8h*)(p + 8);
    return __builtin_shufflevector(lo, hi, 0, 1, 2, 3, 4, 5, 6, 7,
                                           8, 9, 10, 11, 12, 13, 14, 15);
}
// two disjoint 8-f16 runs -> v16h
__device__ inline v16h ld2x8(const _Float16* p0, const _Float16* p1) {
    v8h lo = *(const v8h*)p0;
    v8h hi = *(const v8h*)p1;
    return __builtin_shufflevector(lo, hi, 0, 1, 2, 3, 4, 5, 6, 7,
                                           8, 9, 10, 11, 12, 13, 14, 15);
}

// ---------------------------------------------------------------------------
// small helper kernels
// ---------------------------------------------------------------------------
__global__ __launch_bounds__(256) void cvt_f32_to_f16(const float* __restrict__ in,
                                                      _Float16* __restrict__ out, long n) {
    long i = (long)blockIdx.x * 256 + threadIdx.x;
    if (i < n) out[i] = (_Float16)in[i];
}

// adj -> f16 and transposed f16
__global__ __launch_bounds__(256) void cvt_adj(const float* __restrict__ adj,
                                               _Float16* __restrict__ a_h,
                                               _Float16* __restrict__ aT_h) {
    int b = blockIdx.z, i = blockIdx.y;
    int j = blockIdx.x * 256 + threadIdx.x;
    long base = (long)b * Ss * Ss;
    float v = adj[base + (long)i * Ss + j];
    a_h [base + (long)i * Ss + j] = (_Float16)v;
    aT_h[base + (long)j * Ss + i] = (_Float16)v;
}

// rel_pos int64 -> combined uint8 index: comb[b][q][k] = rp[b][q][k] + 10*rp[b][k][q]
__global__ __launch_bounds__(256) void pack_comb(const long long* __restrict__ rp,
                                                 unsigned char* __restrict__ comb) {
    __shared__ unsigned char bt[32][33];
    const int b = blockIdx.z;
    const int q0 = blockIdx.y * 32, k0 = blockIdx.x * 32;
    const long base = (long)b * Ss * Ss;
    const int col = threadIdx.x & 31, r0 = threadIdx.x >> 5;   // 8 rows / pass
#pragma unroll
    for (int rr = 0; rr < 4; ++rr) {
        int row = rr * 8 + r0;
        bt[row][col] = (unsigned char)rp[base + (long)(k0 + row) * Ss + q0 + col];
    }
    __syncthreads();
#pragma unroll
    for (int rr = 0; rr < 4; ++rr) {
        int row = rr * 8 + r0;
        unsigned char a = (unsigned char)rp[base + (long)(q0 + row) * Ss + k0 + col];
        comb[base + (long)(q0 + row) * Ss + k0 + col] =
            (unsigned char)(a + 10 * bt[col][row]);
    }
}

// row LayerNorm (D=256, one row per 256-thread block) -> f16
__global__ __launch_bounds__(256) void ln_to_h(const float* __restrict__ X,
                                               const float* __restrict__ g,
                                               const float* __restrict__ beta,
                                               _Float16* __restrict__ out) {
    __shared__ float red[256];
    long row = blockIdx.x;
    int  t   = threadIdx.x;
    float v  = X[row * Dd + t];
    red[t] = v; __syncthreads();
    for (int s = 128; s > 0; s >>= 1) { if (t < s) red[t] += red[t + s]; __syncthreads(); }
    float mean = red[0] * (1.0f / Dd);
    __syncthreads();
    float d = v - mean;
    red[t] = d * d; __syncthreads();
    for (int s = 128; s > 0; s >>= 1) { if (t < s) red[t] += red[t + s]; __syncthreads(); }
    float var = red[0] * (1.0f / Dd);
    out[row * Dd + t] = (_Float16)(d * rsqrtf(var + 1e-5f) * g[t] + beta[t]);
}

// ---------------------------------------------------------------------------
// generic LDS-tiled WMMA GEMM: C[M,N] = A[M,K](f16) x B[K,N](f16) + epilogue
// block = 256 threads = 8 waves in a 4(M) x 2(N) arrangement; tile 64x32, K-step 32
// B tile is stored TRANSPOSED in LDS (sBt[n][k], padded rows) so B-fragment
// reads are two ds_load_b128 instead of 16 strided ds_load_u16.
// EP: 0 = f16 out (+bias)          1 = scatter f16 to [B,H,S,32] (+bias)
//     2 = fp32 out = Res + v       3 = fp32 out = v
//     4 = f16 out = relu(Res + v)  5 = scatter f16 to [B,H,32,S] (V transposed)
// ---------------------------------------------------------------------------
template <int EP>
__global__ __launch_bounds__(256) void gemm_wmma(
        const _Float16* __restrict__ A, long strideA,
        const _Float16* __restrict__ Bm, int rowOfsB,
        const float* __restrict__ bias,
        const float* __restrict__ Res, int ldr,
        float* __restrict__ Cf, _Float16* __restrict__ Ch, int ldc,
        int K, int lda, int ldb, int rowOfsC) {
    __shared__ _Float16 sA[64 * 32];
    __shared__ _Float16 sBt[32 * 40];     // [n][k], row stride 40 (80B, 16B-aligned)
    const int z  = blockIdx.z;
    const int m0 = blockIdx.y * 64, n0 = blockIdx.x * 32;
    const int t = threadIdx.x;
    const int lane = t & 31, wv = t >> 5;
    const int nlo = lane & 15, half = lane >> 4;
    const int wm = wv & 3, wn = wv >> 2;

    const _Float16* Ab = A + (long)z * strideA;
    const long brow0 = (long)z * rowOfsB;

    const int ra = t >> 2, ca = (t & 3) * 8;  // A tile: 64 rows x 4 chunks of 8 f16
    v8f acc = {};

    for (int k0 = 0; k0 < K; k0 += 32) {
        *(v8h*)(sA + ra * 32 + ca) =
            *(const v8h*)(Ab + (long)(m0 + ra) * lda + k0 + ca);
        if (t < 128) {
            int rb = t >> 2, cb = (t & 3) * 8;
            v8h bv8 = *(const v8h*)(Bm + (brow0 + k0 + rb) * (long)ldb + n0 + cb);
#pragma unroll
            for (int i = 0; i < 8; ++i) sBt[(cb + i) * 40 + rb] = bv8[i];
        }
        if (k0 + 32 < K)  // global_prefetch_b8 of next A tile
            __builtin_prefetch(Ab + (long)(m0 + ra) * lda + k0 + 32 + ca, 0, 1);
        __syncthreads();

        const _Float16* pa = sA + (wm * 16 + nlo) * 32;
        v16h a  = ld16(pa + 16 * half);
        v16h bf = ld16(sBt + (wn * 16 + nlo) * 40 + 16 * half);

        acc = WMMA_F32_F16(a, bf, acc);
        __syncthreads();
    }

    const int gcol = n0 + wn * 16 + nlo;
    const float bv = bias ? bias[gcol] : 0.f;
#pragma unroll
    for (int r = 0; r < 8; ++r) {
        long grow = (long)z * rowOfsC + m0 + wm * 16 + r + 8 * half;
        float v = acc[r] + bv;
        if constexpr (EP == 0) {
            Ch[grow * ldc + gcol] = (_Float16)v;
        } else if constexpr (EP == 1) {
            int bb = (int)(grow >> 10), s = (int)(grow & 1023);
            int hh = gcol >> 5, d = gcol & 31;
            Ch[(((long)bb * Hh + hh) * Ss + s) * DK + d] = (_Float16)v;
        } else if constexpr (EP == 2) {
            Cf[grow * ldc + gcol] = Res[grow * (long)ldr + gcol] + v;
        } else if constexpr (EP == 3) {
            Cf[grow * ldc + gcol] = v;
        } else if constexpr (EP == 4) {
            float u = v + Res[grow * (long)ldr + gcol];
            Ch[grow * ldc + gcol] = (_Float16)fmaxf(u, 0.f);
        } else {  // EP == 5: V stored transposed [B,H,32,S]
            int bb = (int)(grow >> 10), s = (int)(grow & 1023);
            int hh = gcol >> 5, d = gcol & 31;
            Ch[(((long)bb * Hh + hh) * DK + d) * Ss + s] = (_Float16)v;
        }
    }
}

// ---------------------------------------------------------------------------
// flash attention (fixed-shift variant): one block = (b, h, 16-query tile);
// 8 waves, each owning a 128-column K/V strip processed as 4 chunks of 32.
// Pre-LN encoder scores are O(1), so unshifted exp2 has huge fp32 headroom:
// no running max, no per-chunk cross-lane reductions, no rescales. log2e is
// folded into the bias LUT so the activation is a bare v_exp_f32.
// V is consumed from a transposed [B,H,32,S] layout -> contiguous fragments.
// ---------------------------------------------------------------------------
__global__ __launch_bounds__(256) void attn_flash(
        const _Float16* __restrict__ Q,   // [B,H,S,32]
        const _Float16* __restrict__ Kt,  // [B,H,S,32]
        const _Float16* __restrict__ Vt,  // [B,H,32,S] (transposed)
        const unsigned char* __restrict__ comb,  // [B,S,S] idx = f + 10*b
        const float* __restrict__ embF,   // [10,H]
        const float* __restrict__ embB,   // [10,H]
        _Float16* __restrict__ O) {       // [B,S,256]
    __shared__ _Float16 pbuf[8][16 * 32];
    __shared__ float Obuf[8][16][32];
    __shared__ float Lbuf[8][16];
    __shared__ float lut[100];

    const int qt = blockIdx.x, h = blockIdx.y, b = blockIdx.z;
    const int q0 = qt * 16;
    const int t = threadIdx.x;
    const int lane = t & 31, wv = t >> 5;
    const int nlo = lane & 15, half = lane >> 4;

    const float scale_l2e = 0.17677669529663688f * 1.4426950408889634f; // log2e/sqrt(32)
    if (t < 100) {
        int fi = t % 10, bi = t / 10;
        lut[t] = scale_l2e * (1.f + embF[fi * Hh + h] + embB[bi * Hh + h]);
    }

    const _Float16* qb = Q  + (((long)b * Hh + h) * Ss) * DK;
    const _Float16* kb = Kt + (((long)b * Hh + h) * Ss) * DK;
    const _Float16* vt = Vt + (((long)b * Hh + h) * DK) * Ss;
    const unsigned char* cbp = comb + (long)b * Ss * Ss;

    // q A-fragment (16x32, reused for all k chunks)
    const _Float16* qp = qb + (long)(q0 + nlo) * DK;
    v16h aq = ld16(qp + 16 * half);

    float Lpart[8];                 // per-lane partial row sums (no shuffles in loop)
    v8f O0 = {}, O1 = {};
#pragma unroll
    for (int r = 0; r < 8; ++r) Lpart[r] = 0.f;

    __syncthreads();   // lut ready

    for (int it = 0; it < 4; ++it) {
        const int c0 = (wv * 4 + it) * 32;

        // K^T B-fragments: B[d][n] = k[c0+n][d]  (16 contiguous f16 each)
        v16h bk0 = ld16(kb + (long)(c0 + nlo) * DK + 16 * half);
        v16h bk1 = ld16(kb + (long)(c0 + 16 + nlo) * DK + 16 * half);
        v8f s0 = {}, s1 = {};
        s0 = WMMA_F32_F16(aq, bk0, s0);
        s1 = WMMA_F32_F16(aq, bk1, s1);

        // fused scale + multiplicative rel-pos bias (log2e folded), bare exp2
#pragma unroll
        for (int r = 0; r < 8; ++r) {
            int qi = q0 + r + 8 * half;
            long rowo = (long)qi * Ss + c0 + nlo;
            float p0 = __builtin_amdgcn_exp2f(s0[r] * lut[cbp[rowo]]);
            float p1 = __builtin_amdgcn_exp2f(s1[r] * lut[cbp[rowo + 16]]);
            s0[r] = p0; s1[r] = p1;
            Lpart[r] += p0 + p1;
        }

        // bounce p through LDS to convert C-layout -> A-layout (per-wave region)
        _Float16* pw = pbuf[wv];
#pragma unroll
        for (int r = 0; r < 8; ++r) {
            pw[(r + 8 * half) * 32 + nlo]      = (_Float16)s0[r];
            pw[(r + 8 * half) * 32 + 16 + nlo] = (_Float16)s1[r];
        }
        __builtin_amdgcn_wave_barrier();   // LDS ops are in-order per wave

        const _Float16* pp = pw + nlo * 32;
        v16h ap = ld16(pp + 16 * half);

        // V B-fragments from transposed layout: B[k][d] = vt[d][k] (contiguous)
        v16h bv0 = ld16(vt + (long)nlo * Ss + c0 + 16 * half);
        v16h bv1 = ld16(vt + (long)(16 + nlo) * Ss + c0 + 16 * half);
        O0 = WMMA_F32_F16(ap, bv0, O0);
        O1 = WMMA_F32_F16(ap, bv1, O1);
    }

    // one-time 16-lane reduction of the per-lane row sums
#pragma unroll
    for (int r = 0; r < 8; ++r) {
        float v = Lpart[r];
        for (int off = 1; off < 16; off <<= 1) v += __shfl_xor(v, off, 32);
        Lpart[r] = v;
    }

    // merge 8 per-wave partials (plain sums)
#pragma unroll
    for (int r = 0; r < 8; ++r) {
        Obuf[wv][r + 8 * half][nlo]      = O0[r];
        Obuf[wv][r + 8 * half][16 + nlo] = O1[r];
    }
    if (nlo == 0) {
#pragma unroll
        for (int r = 0; r < 8; ++r) Lbuf[wv][r + 8 * half] = Lpart[r];
    }
    __syncthreads();

    for (int e = t; e < 512; e += 256) {
        int row = e >> 5, col = e & 31;
        float den = 0.f, num = 0.f;
#pragma unroll
        for (int w = 0; w < 8; ++w) {
            den += Lbuf[w][row];
            num += Obuf[w][row][col];
        }
        O[((long)b * Ss + q0 + row) * Dd + h * DK + col] = (_Float16)(num / den);
    }
}

// ---------------------------------------------------------------------------
// launch
// ---------------------------------------------------------------------------
extern "C" void kernel_launch(void* const* d_in, const int* in_sizes, int n_in,
                              void* d_out, int out_size, void* d_ws, size_t ws_size,
                              hipStream_t stream) {
    (void)in_sizes; (void)n_in; (void)out_size; (void)ws_size;
    const float*     x      = (const float*)d_in[0];
    const long long* relpos = (const long long*)d_in[1];
    const float*     adj    = (const float*)d_in[2];
    const float *ln1_g = (const float*)d_in[3],  *ln1_b = (const float*)d_in[4];
    const float *Wk = (const float*)d_in[5],  *bk = (const float*)d_in[6];
    const float *Wv = (const float*)d_in[7],  *bv = (const float*)d_in[8];
    const float *Wq = (const float*)d_in[9],  *bq = (const float*)d_in[10];
    const float *Wo = (const float*)d_in[11], *bo = (const float*)d_in[12];
    const float *embF = (const float*)d_in[13], *embB = (const float*)d_in[14];
    const float *ln2_g = (const float*)d_in[15], *ln2_b = (const float*)d_in[16];
    const float *W1 = (const float*)d_in[17], *b1 = (const float*)d_in[18];
    const float *Wg = (const float*)d_in[19], *bg = (const float*)d_in[20];
    const float *W2 = (const float*)d_in[21], *b2 = (const float*)d_in[22];
    float* out = (float*)d_out;

    char* ws = (char*)d_ws;
    size_t off = 0;
    auto alloc = [&](size_t bytes) -> char* {
        char* p = ws + off;
        off = (off + bytes + 255) & ~(size_t)255;
        return p;
    };
    _Float16* xn1_h = (_Float16*)alloc((size_t)Mrows * Dd * 2);
    _Float16* q_h   = (_Float16*)alloc((size_t)Mrows * Dd * 2);
    _Float16* k_h   = (_Float16*)alloc((size_t)Mrows * Dd * 2);
    _Float16* vt_h  = (_Float16*)alloc((size_t)Mrows * Dd * 2);   // [B,H,32,S]
    _Float16* o_h   = (_Float16*)alloc((size_t)Mrows * Dd * 2);
    _Float16* Wq_h  = (_Float16*)alloc((size_t)Dd * Dd * 2);
    _Float16* Wk_h  = (_Float16*)alloc((size_t)Dd * Dd * 2);
    _Float16* Wv_h  = (_Float16*)alloc((size_t)Dd * Dd * 2);
    _Float16* Wo_h  = (_Float16*)alloc((size_t)Dd * Dd * 2);
    _Float16* W1_h  = (_Float16*)alloc((size_t)Dd * HID * 2);
    _Float16* Wg_h  = (_Float16*)alloc((size_t)Dd * HID * 2);
    _Float16* W2_h  = (_Float16*)alloc((size_t)HID * Dd * 2);
    unsigned char* comb = (unsigned char*)alloc((size_t)Bb * Ss * Ss);
    _Float16* adj_h  = (_Float16*)alloc((size_t)Bb * Ss * Ss * 2);
    _Float16* adjT_h = (_Float16*)alloc((size_t)Bb * Ss * Ss * 2);
    float*    x2     = (float*)alloc((size_t)Mrows * Dd * 4);
    _Float16* xn2_h  = (_Float16*)alloc((size_t)Mrows * Dd * 2);
    float*    x1     = (float*)alloc((size_t)Mrows * HID * 4);
    _Float16* g_h    = (_Float16*)alloc((size_t)Mrows * HID * 2);
    _Float16* h_h    = (_Float16*)alloc((size_t)Mrows * HID * 2);

    // --- precision conversion / packing ---
    cvt_f32_to_f16<<<dim3(Dd * Dd / 256), 256, 0, stream>>>(Wq, Wq_h, (long)Dd * Dd);
    cvt_f32_to_f16<<<dim3(Dd * Dd / 256), 256, 0, stream>>>(Wk, Wk_h, (long)Dd * Dd);
    cvt_f32_to_f16<<<dim3(Dd * Dd / 256), 256, 0, stream>>>(Wv, Wv_h, (long)Dd * Dd);
    cvt_f32_to_f16<<<dim3(Dd * Dd / 256), 256, 0, stream>>>(Wo, Wo_h, (long)Dd * Dd);
    cvt_f32_to_f16<<<dim3(Dd * HID / 256), 256, 0, stream>>>(W1, W1_h, (long)Dd * HID);
    cvt_f32_to_f16<<<dim3(Dd * HID / 256), 256, 0, stream>>>(Wg, Wg_h, (long)Dd * HID);
    cvt_f32_to_f16<<<dim3(HID * Dd / 256), 256, 0, stream>>>(W2, W2_h, (long)HID * Dd);
    pack_comb<<<dim3(Ss / 32, Ss / 32, Bb), 256, 0, stream>>>(relpos, comb);
    cvt_adj<<<dim3(Ss / 256, Ss, Bb), 256, 0, stream>>>(adj, adj_h, adjT_h);

    // --- attention block ---
    ln_to_h<<<dim3(Mrows), 256, 0, stream>>>(x, ln1_g, ln1_b, xn1_h);
    dim3 gP(Dd / 32, Mrows / 64, 1);
    gemm_wmma<1><<<gP, 256, 0, stream>>>(xn1_h, 0, Wq_h, 0, bq, nullptr, 0, nullptr, q_h, 0, Dd, Dd, Dd, 0);
    gemm_wmma<1><<<gP, 256, 0, stream>>>(xn1_h, 0, Wk_h, 0, bk, nullptr, 0, nullptr, k_h, 0, Dd, Dd, Dd, 0);
    gemm_wmma<5><<<gP, 256, 0, stream>>>(xn1_h, 0, Wv_h, 0, bv, nullptr, 0, nullptr, vt_h, 0, Dd, Dd, Dd, 0);
    attn_flash<<<dim3(Ss / 16, Hh, Bb), 256, 0, stream>>>(q_h, k_h, vt_h, comb, embF, embB, o_h);
    gemm_wmma<2><<<gP, 256, 0, stream>>>(o_h, 0, Wo_h, 0, bo, x, Dd, x2, nullptr, Dd, Dd, Dd, Dd, 0);

    // --- GCN feed-forward block ---
    ln_to_h<<<dim3(Mrows), 256, 0, stream>>>(x2, ln2_g, ln2_b, xn2_h);
    dim3 gH(HID / 32, Mrows / 64, 1);
    gemm_wmma<3><<<gH, 256, 0, stream>>>(xn2_h, 0, W1_h, 0, b1, nullptr, 0, x1, nullptr, HID, Dd, Dd, HID, 0);
    gemm_wmma<0><<<gH, 256, 0, stream>>>(xn2_h, 0, Wg_h, 0, bg, nullptr, 0, nullptr, g_h, HID, Dd, Dd, HID, 0);
    // h[:, :512]  = relu(x1[:, :512]  + adj  @ g1)   (batched over b)
    // h[:, 512:]  = relu(x1[:, 512:]  + adjT @ g2)
    dim3 gG(512 / 32, Ss / 64, Bb);
    gemm_wmma<4><<<gG, 256, 0, stream>>>(adj_h,  (long)Ss * Ss, g_h,       Ss, nullptr, x1,       HID,
                                         nullptr, h_h,       HID, Ss, Ss, HID, Ss);
    gemm_wmma<4><<<gG, 256, 0, stream>>>(adjT_h, (long)Ss * Ss, g_h + 512, Ss, nullptr, x1 + 512, HID,
                                         nullptr, h_h + 512, HID, Ss, Ss, HID, Ss);
    // out = x2 + h @ W2 + b2
    gemm_wmma<2><<<gP, 256, 0, stream>>>(h_h, 0, W2_h, 0, b2, x2, Dd, out, nullptr, Dd, HID, HID, Dd, 0);
}
